// LSTM_Att_39479339385341
// MI455X (gfx1250) — matrix operand order repaired and checked by reading, not compiled, not run on hardware
//
#include <hip/hip_runtime.h>

// ---------------- problem constants ----------------
#define Bc 256
#define Tc 64
#define Dc 1024
#define Hc 1024
#define Ac 512
#define Cc 128

typedef __attribute__((ext_vector_type(16))) __bf16 v16bf;
typedef __attribute__((ext_vector_type(8)))  float  v8f;
typedef __attribute__((ext_vector_type(8)))  unsigned short v8us;

union V16 { v16bf v; v8us h[2]; };

__device__ inline unsigned short f2bf(float f) {
    unsigned u = __builtin_bit_cast(unsigned, f);
    unsigned r = u + 0x7FFFu + ((u >> 16) & 1u);   // round-to-nearest-even
    return (unsigned short)(r >> 16);
}
__device__ inline float bf2f(unsigned short s) {
    return __builtin_bit_cast(float, ((unsigned)s) << 16);
}
__device__ inline float sigm(float x) { return 1.f / (1.f + __expf(-x)); }

// ---------------- bf16 conversion (rows with strides) ----------------
__global__ void k_cvt(const float* __restrict__ src, int ldsrc,
                      unsigned short* __restrict__ dst, int lddst, int dstoff,
                      int rows, int cols) {
    int idx = blockIdx.x * blockDim.x + threadIdx.x;
    if (idx >= rows * cols) return;
    int r = idx / cols, c = idx % cols;
    dst[(size_t)r * lddst + dstoff + c] = f2bf(src[(size_t)r * ldsrc + c]);
}

__global__ void k_badd(float* __restrict__ dst, const float* __restrict__ a,
                       const float* __restrict__ b, int n) {
    int i = blockIdx.x * blockDim.x + threadIdx.x;
    if (i < n) dst[i] = a[i] + b[i];
}

__global__ void k_init(float* __restrict__ c, unsigned short* __restrict__ s_bf, int n) {
    int i = blockIdx.x * blockDim.x + threadIdx.x;
    if (i < n) { c[i] = 0.f; s_bf[i] = 0; }
}

// ---------------- WMMA GEMM: out[m,n] = sum_seg A_seg[m,:]·W_seg[n,:] + bias + gather ----------------
// Each wave computes a 16(M) x 64(N) tile via 4 accumulators (A fragment reused 4x).
__device__ inline void gemm_seg(v8f (&acc)[4],
                                const unsigned short* __restrict__ A, int lda,
                                const unsigned short* __restrict__ W, int ldw, int K,
                                int m0, int n0, int lane) {
    int mr = lane & 15, half = lane >> 4;
    const unsigned short* arow = A + (size_t)(m0 + mr) * lda + 8 * half;
    for (int k = 0; k < K; k += 32) {
        V16 af;
        af.h[0] = *(const v8us*)(arow + k);        // K = 8h .. 8h+7
        af.h[1] = *(const v8us*)(arow + k + 16);   // K = 16+8h .. +7
#pragma unroll
        for (int nn = 0; nn < 4; nn++) {
            const unsigned short* wrow =
                W + (size_t)(n0 + nn * 16 + mr) * ldw + k + 8 * half;
            V16 bf;
            bf.h[0] = *(const v8us*)(wrow);        // K = 8h .. 8h+7
            bf.h[1] = *(const v8us*)(wrow + 16);   // K = 16+8h .. +7
            acc[nn] = __builtin_amdgcn_wmma_f32_16x16x32_bf16(
                false, af.v, false, bf.v, (short)0, acc[nn], false, false);
        }
    }
}

__global__ void __launch_bounds__(128)
k_gemm(const unsigned short* __restrict__ A0, int lda0, int K0,
       const unsigned short* __restrict__ W0, int ldw0,
       const unsigned short* __restrict__ A1, int lda1, int K1,
       const unsigned short* __restrict__ W1, int ldw1,
       const float* __restrict__ bias,
       const float* __restrict__ Wg, int ldg, int gbase,
       const int* __restrict__ labels, int t,
       float* __restrict__ outp, int ldout, int M, int N) {
    int wave = blockIdx.x * (blockDim.x >> 5) + (threadIdx.x >> 5);
    int lane = threadIdx.x & 31;
    int mtiles = M >> 4;
    int m0 = (wave % mtiles) << 4;
    int n0 = (wave / mtiles) << 6;
    if (n0 >= N) return;

    v8f acc[4] = {};
    gemm_seg(acc, A0, lda0, W0, ldw0, K0, m0, n0, lane);
    if (K1 > 0) gemm_seg(acc, A1, lda1, W1, ldw1, K1, m0, n0, lane);

    int mr = lane & 15, half = lane >> 4;
#pragma unroll
    for (int nn = 0; nn < 4; nn++) {
        int n = n0 + nn * 16 + mr;
        float bia = bias ? bias[n] : 0.f;
#pragma unroll
        for (int r = 0; r < 8; r++) {
            int m = m0 + r + 8 * half;
            float val = acc[nn][r] + bia;
            if (labels) {   // teacher-forcing one-hot => gather one weight column
                int tf = (t == 0) ? 0 : labels[m * Tc + t - 1];
                val += Wg[(size_t)n * ldg + gbase + tf];
            }
            outp[(size_t)m * ldout + n] = val;
        }
    }
}

// ---------------- attention: scores -> softmax over T -> context gt ----------------
__global__ void __launch_bounds__(256)
k_attn(const float* __restrict__ wsbuf,       // [B, A]
       const float* __restrict__ Vh,          // [B*T, A]
       const float* __restrict__ vvec,        // [A]
       const unsigned short* __restrict__ x_bf, // [B*T, D]
       unsigned short* __restrict__ gt_bf) {  // [B, D]
    __shared__ float sws[Ac];
    __shared__ float sv[Ac];
    __shared__ float se[Tc];
    __shared__ float salpha[Tc];
    int b = blockIdx.x, tid = threadIdx.x;
    sws[tid] = wsbuf[b * Ac + tid];
    sws[tid + 256] = wsbuf[b * Ac + 256 + tid];
    sv[tid] = vvec[tid];
    sv[tid + 256] = vvec[tid + 256];
    __syncthreads();

    int wave = tid >> 5, lane = tid & 31;
    for (int t = wave; t < Tc; t += 8) {
        const float* vh = Vh + ((size_t)b * Tc + t) * Ac;
        float p = 0.f;
        for (int a = lane; a < Ac; a += 32)
            p += tanhf(sws[a] + vh[a]) * sv[a];
#pragma unroll
        for (int off = 16; off; off >>= 1) p += __shfl_xor(p, off, 32);
        if (lane == 0) se[t] = p;
    }
    __syncthreads();

    float mx = -1e30f;
    for (int t = 0; t < Tc; t++) mx = fmaxf(mx, se[t]);
    float sum = 0.f;
    for (int t = 0; t < Tc; t++) sum += __expf(se[t] - mx);
    float inv = 1.f / sum;
    if (tid < Tc) salpha[tid] = __expf(se[tid] - mx) * inv;
    __syncthreads();

    for (int d = tid; d < Dc; d += 256) {
        const unsigned short* xp = x_bf + (size_t)b * Tc * Dc + d;
        float acc = 0.f;
#pragma unroll 4
        for (int t = 0; t < Tc; t++) acc += salpha[t] * bf2f(xp[t * Dc]);
        gt_bf[b * Dc + d] = f2bf(acc);
    }
}

// ---------------- LSTM cell pointwise ----------------
__global__ void __launch_bounds__(256)
k_cell(const float* __restrict__ gates, float* __restrict__ cst,
       unsigned short* __restrict__ s_bf) {
    int idx = blockIdx.x * blockDim.x + threadIdx.x;   // B*H
    int b = idx >> 10, h = idx & 1023;
    const float* g = gates + (size_t)b * 4 * Hc;
    float i = sigm(g[h]);
    float f = sigm(g[Hc + h]);
    float gg = tanhf(g[2 * Hc + h]);
    float o = sigm(g[3 * Hc + h]);
    float cn = f * cst[idx] + i * gg;
    cst[idx] = cn;
    s_bf[idx] = f2bf(o * tanhf(cn));
}

// ---------------- host launcher ----------------
extern "C" void kernel_launch(void* const* d_in, const int* in_sizes, int n_in,
                              void* d_out, int out_size, void* d_ws, size_t ws_size,
                              hipStream_t stream) {
    (void)in_sizes; (void)n_in; (void)out_size; (void)ws_size;
    const float* x      = (const float*)d_in[0];
    const int*   labels = (const int*)d_in[1];
    const float* Ws     = (const float*)d_in[2];
    const float* bs     = (const float*)d_in[3];
    const float* Wh     = (const float*)d_in[4];
    const float* bh     = (const float*)d_in[5];
    const float* vvec   = (const float*)d_in[6];
    const float* W_ih   = (const float*)d_in[7];
    const float* W_hh   = (const float*)d_in[8];
    const float* b_ih   = (const float*)d_in[9];
    const float* b_hh   = (const float*)d_in[10];
    const float* W_out1 = (const float*)d_in[11];
    const float* b_out1 = (const float*)d_in[12];
    const float* W_out2 = (const float*)d_in[13];
    const float* b_out2 = (const float*)d_in[14];
    float* out = (float*)d_out;

    char* p = (char*)d_ws;
    auto alloc = [&](size_t bytes) -> void* {
        void* r = (void*)p; p += (bytes + 255) & ~(size_t)255; return r;
    };
    unsigned short* x_bf    = (unsigned short*)alloc((size_t)Bc * Tc * Dc * 2);
    float*          Vh      = (float*)alloc((size_t)Bc * Tc * Ac * 4);
    unsigned short* Ws_bf   = (unsigned short*)alloc((size_t)Ac * Hc * 2);
    unsigned short* Wh_bf   = (unsigned short*)alloc((size_t)Ac * Dc * 2);
    unsigned short* Wcat    = (unsigned short*)alloc((size_t)4 * Hc * 2048 * 2);
    unsigned short* Woutcat = (unsigned short*)alloc((size_t)Cc * 2048 * 2);
    float*          bias_g  = (float*)alloc((size_t)4 * Hc * 4);
    float*          bias_y  = (float*)alloc((size_t)Cc * 4);
    float*          cstate  = (float*)alloc((size_t)Bc * Hc * 4);
    unsigned short* s_bf    = (unsigned short*)alloc((size_t)Bc * Hc * 2);
    unsigned short* gt_bf   = (unsigned short*)alloc((size_t)Bc * Dc * 2);
    float*          wsbuf   = (float*)alloc((size_t)Bc * Ac * 4);
    float*          gates   = (float*)alloc((size_t)Bc * 4 * Hc * 4);

    auto cvt = [&](const float* src, int ldsrc, unsigned short* dst, int lddst,
                   int off, int rows, int cols) {
        int n = rows * cols;
        k_cvt<<<dim3((n + 255) / 256), dim3(256), 0, stream>>>(src, ldsrc, dst, lddst, off, rows, cols);
    };
    cvt(x,      Dc,      x_bf,    Dc,   0,    Bc * Tc, Dc);
    cvt(Ws,     Hc,      Ws_bf,   Hc,   0,    Ac,      Hc);
    cvt(Wh,     Dc,      Wh_bf,   Dc,   0,    Ac,      Dc);
    cvt(W_ih,   Dc + Cc, Wcat,    2048, 0,    4 * Hc,  Dc);
    cvt(W_hh,   Hc,      Wcat,    2048, 1024, 4 * Hc,  Hc);
    cvt(W_out1, Hc,      Woutcat, 2048, 0,    Cc,      Hc);
    cvt(W_out2, Dc + Cc, Woutcat, 2048, 1024, Cc,      Dc);
    k_badd<<<dim3(16), dim3(256), 0, stream>>>(bias_g, b_ih, b_hh, 4 * Hc);
    k_badd<<<dim3(1), dim3(128), 0, stream>>>(bias_y, b_out1, b_out2, Cc);
    k_init<<<dim3((Bc * Hc) / 256), dim3(256), 0, stream>>>(cstate, s_bf, Bc * Hc);

    auto gemm = [&](const unsigned short* A0, int lda0, int K0,
                    const unsigned short* W0, int ldw0,
                    const unsigned short* A1, int lda1, int K1,
                    const unsigned short* W1, int ldw1,
                    const float* bias, const float* Wg, int ldg, int gbase,
                    const int* lab, int t, float* outp, int ldout, int M, int N) {
        int waves = (M / 16) * (N / 64);
        k_gemm<<<dim3(waves / 4), dim3(128), 0, stream>>>(
            A0, lda0, K0, W0, ldw0, A1, lda1, K1, W1, ldw1,
            bias, Wg, ldg, gbase, lab, t, outp, ldout, M, N);
    };

    // Vh = x @ Wh^T + bh   (one-time, M = B*T)
    gemm(x_bf, Dc, Dc, Wh_bf, Dc, nullptr, 0, 0, nullptr, 0,
         bh, nullptr, 0, 0, nullptr, 0, Vh, Ac, Bc * Tc, Ac);

    for (int t = 0; t < Tc; t++) {
        // ws = s @ Ws^T + bs
        gemm(s_bf, Hc, Hc, Ws_bf, Hc, nullptr, 0, 0, nullptr, 0,
             bs, nullptr, 0, 0, nullptr, 0, wsbuf, Ac, Bc, Ac);
        // attention scores + softmax + context
        k_attn<<<dim3(Bc), dim3(256), 0, stream>>>(wsbuf, Vh, vvec, x_bf, gt_bf);
        // gates = gt@W_ih[:, :D]^T + s@W_hh^T + bias + W_ih[:, D+tf]
        gemm(gt_bf, Dc, Dc, Wcat, 2048, s_bf, Hc, Hc, Wcat + 1024, 2048,
             bias_g, W_ih, Dc + Cc, Dc, labels, t, gates, 4 * Hc, Bc, 4 * Hc);
        // LSTM pointwise: updates cstate, writes s_new into s_bf
        k_cell<<<dim3((Bc * Hc) / 256), dim3(256), 0, stream>>>(gates, cstate, s_bf);
        // y = s_new@W_out1^T + gt@W_out2[:, :D]^T + biases + W_out2[:, D+tf]
        gemm(s_bf, Hc, Hc, Woutcat, 2048, gt_bf, Dc, Dc, Woutcat + 1024, 2048,
             bias_y, W_out2, Dc + Cc, Dc, labels, t, out + (size_t)t * Cc, Tc * Cc, Bc, Cc);
    }
}
